// MicroAdder_16501264351743
// MI455X (gfx1250) — compile-verified, matrix operand order checked by reading, not run on hardware
//
#include <hip/hip_runtime.h>
#include <hip/hip_bf16.h>

typedef _Float16 h8   __attribute__((ext_vector_type(8)));
typedef _Float16 v16h __attribute__((ext_vector_type(16)));
typedef float    v8f  __attribute__((ext_vector_type(8)));
typedef int      v4i  __attribute__((ext_vector_type(4)));

#define D_MODEL 1024
#define T_SEQ   1024
#define BATCH   4
#define NHEAD   16
#define KVHEADS 4
#define HEADD   64
#define FFN_DIM 4096
#define M_ROWS  4096   // BATCH*T_SEQ

// --------------------------------------------------------------------------
// 16-byte global->LDS copy. Use the CDNA5 async LDS-DMA path
// (GLOBAL_LOAD_ASYNC_TO_LDS_B128, tracked by ASYNCcnt) when the toolchain
// declares the builtin; otherwise stage through VGPRs (global_load_b128 +
// ds_store_b128). Builtin signature (from hipcc diagnostic): arg0 is
// v4i addrspace(1)*, arg1 v4i addrspace(3)*, then imm offset / imm cpol.
// --------------------------------------------------------------------------
#if defined(__has_builtin)
#if __has_builtin(__builtin_amdgcn_global_load_async_to_lds_b128)
#define USE_ASYNC_LDS 1
#endif
#endif

#ifdef USE_ASYNC_LDS
typedef __attribute__((address_space(1))) v4i* as1_v4i;
typedef __attribute__((address_space(3))) v4i* as3_v4i;
#endif

__device__ __forceinline__ void cp16(_Float16* lds, const _Float16* g) {
#ifdef USE_ASYNC_LDS
  __builtin_amdgcn_global_load_async_to_lds_b128((as1_v4i)g, (as3_v4i)lds, 0, 0);
#else
  *(h8*)lds = *(const h8*)g;
#endif
}

__device__ __forceinline__ void async_fence() {
#ifdef USE_ASYNC_LDS
#if defined(__has_builtin) && __has_builtin(__builtin_amdgcn_s_wait_asynccnt)
  __builtin_amdgcn_s_wait_asynccnt(0);
#else
  asm volatile("s_wait_asynccnt 0" ::: "memory");
#endif
#endif
}

__device__ __forceinline__ v8f vzero8() {
  v8f z;
#pragma unroll
  for (int i = 0; i < 8; i++) z[i] = 0.f;
  return z;
}

// Load one 16x32 f16 WMMA A/B fragment from an LDS tile (row-major, stride ld halves).
// Per CDNA5 ISA 16-bit A layout: lane L -> row L%16; half = L/16;
// VGPR0..3 = K (half*8 + 0..7), VGPR4..7 = K (16 + half*8 + 0..7).
__device__ __forceinline__ v16h ldfrag(const _Float16* base, int ld) {
  const int lane = threadIdx.x & 31;
  const int row  = lane & 15;
  const int hsel = lane >> 4;
  const _Float16* p0 = base + row * ld + hsel * 8;
  v16h f;
#pragma unroll
  for (int i = 0; i < 8; i++) f[i] = p0[i];
#pragma unroll
  for (int i = 0; i < 8; i++) f[8 + i] = p0[16 + i];
  return f;
}

// ---------------------------------------------------------------------------
// Tiled WMMA GEMM: C[M,N] = A[M,K](f16, stride lda) @ Bt[N,K](f16)^T
// Epilogue selected at compile time: +bias, exact GELU, +residual(f32),
// f32 or f16 output. Grid: (N/128, M/128), 256 threads, BM=BN=128, BK=64.
// ---------------------------------------------------------------------------
template <bool HAS_BIAS, bool DO_GELU, bool HAS_RES, bool OUT_F16>
__global__ void __launch_bounds__(256)
gemm_f16(const _Float16* __restrict__ A, int lda,
         const _Float16* __restrict__ Bt, int K, int N,
         const float* __restrict__ bias, const float* __restrict__ residual,
         float* __restrict__ outF, _Float16* __restrict__ outH) {
  __shared__ _Float16 As[128 * 72];
  __shared__ _Float16 Bs[128 * 72];
  const int tid  = threadIdx.x;
  const int wave = tid >> 5, lane = tid & 31;
  const int wm   = wave >> 1, wn = wave & 1;
  const int hsel = lane >> 4, ln = lane & 15;
  const int bm = blockIdx.y * 128, bn = blockIdx.x * 128;

  v8f acc[2][4];
#pragma unroll
  for (int i = 0; i < 2; i++)
#pragma unroll
    for (int j = 0; j < 4; j++) acc[i][j] = vzero8();

  for (int kb = 0; kb < K; kb += 64) {
#pragma unroll
    for (int i = 0; i < 4; i++) {
      int v = tid + i * 256;
      int r = v >> 3, c = (v & 7) * 8;
      cp16(As + r * 72 + c, A + (size_t)(bm + r) * lda + kb + c);
      cp16(Bs + r * 72 + c, Bt + (size_t)(bn + r) * K + kb + c);
    }
    async_fence();
    __syncthreads();
#pragma unroll
    for (int kk = 0; kk < 2; kk++) {
      v16h af[2];
#pragma unroll
      for (int i = 0; i < 2; i++)
        af[i] = ldfrag(As + (wm * 32 + i * 16) * 72 + kk * 32, 72);
#pragma unroll
      for (int j = 0; j < 4; j++) {
        v16h bf = ldfrag(Bs + (wn * 64 + j * 16) * 72 + kk * 32, 72);
#pragma unroll
        for (int i = 0; i < 2; i++)
          acc[i][j] = __builtin_amdgcn_wmma_f32_16x16x32_f16(
              false, af[i], false, bf, (short)0, acc[i][j], false, false);
      }
    }
    __syncthreads();
  }

#pragma unroll
  for (int i = 0; i < 2; i++) {
#pragma unroll
    for (int j = 0; j < 4; j++) {
#pragma unroll
      for (int r = 0; r < 8; r++) {
        int row = bm + wm * 32 + i * 16 + r + 8 * hsel;
        int col = bn + wn * 64 + j * 16 + ln;
        float v = acc[i][j][r];
        if (HAS_BIAS) v += bias[col];
        if (DO_GELU) v = 0.5f * v * (1.f + erff(v * 0.70710678118f));
        if (HAS_RES) v += residual[(size_t)row * N + col];
        if (OUT_F16) outH[(size_t)row * N + col] = (_Float16)v;
        else         outF[(size_t)row * N + col] = v;
      }
    }
  }
}

// ---------------------------------------------------------------------------
// Flash attention (causal + ALiBi + GQA). Grid: (T/128, B*H), 256 threads.
// Wave w owns 16 complete query rows -> softmax reductions stay in a 16-lane
// half-wave (wave32). QK^T and PV via WMMA f16->f32. Scale folded into Q.
// ---------------------------------------------------------------------------
__global__ void __launch_bounds__(256)
flash_attn(const _Float16* __restrict__ q16, const _Float16* __restrict__ k16,
           const _Float16* __restrict__ vt16, const float* __restrict__ alibi,
           _Float16* __restrict__ o16) {
  __shared__ _Float16 Qs[128 * 72];
  __shared__ _Float16 Ks[64 * 72];
  __shared__ _Float16 Vs[64 * 72];
  __shared__ _Float16 Ps[8 * 16 * 72];
  const int tid  = threadIdx.x;
  const int wave = tid >> 5, lane = tid & 31;
  const int hsel = lane >> 4, ln = lane & 15;
  const int bh = blockIdx.y;
  const int b = bh >> 4, h = bh & 15;
  const int kv = b * KVHEADS + (h >> 2);  // REP = 4
  const int qBlock = blockIdx.x * 128;
  const float slope = __expf(alibi[h]);

#pragma unroll
  for (int i = 0; i < 4; i++) {
    int v = tid + i * 256;
    int r = v >> 3, c = (v & 7) * 8;
    cp16(Qs + r * 72 + c, q16 + ((size_t)bh * T_SEQ + qBlock + r) * HEADD + c);
  }
  async_fence();
  __syncthreads();
  v16h aQ[2];
#pragma unroll
  for (int kk = 0; kk < 2; kk++) aQ[kk] = ldfrag(Qs + (wave * 16) * 72 + kk * 32, 72);

  v8f oacc[4];
#pragma unroll
  for (int j = 0; j < 4; j++) oacc[j] = vzero8();
  float m_arr[8], l_arr[8];
#pragma unroll
  for (int r = 0; r < 8; r++) { m_arr[r] = -1e30f; l_arr[r] = 0.f; }
  const int qBase = qBlock + wave * 16;
  const int ktEnd = blockIdx.x * 2 + 2;

  for (int kt = 0; kt < ktEnd; kt++) {
    __syncthreads();
#pragma unroll
    for (int i = 0; i < 2; i++) {
      int v = tid + i * 256;
      int r = v >> 3, c = (v & 7) * 8;
      cp16(Ks + r * 72 + c, k16 + ((size_t)kv * T_SEQ + kt * 64 + r) * HEADD + c);
      cp16(Vs + r * 72 + c, vt16 + ((size_t)kv * HEADD + r) * T_SEQ + kt * 64 + c);
    }
    async_fence();
    __syncthreads();

    v8f sacc[4];
#pragma unroll
    for (int j = 0; j < 4; j++) sacc[j] = vzero8();
#pragma unroll
    for (int kk = 0; kk < 2; kk++) {
#pragma unroll
      for (int j = 0; j < 4; j++) {
        v16h bK = ldfrag(Ks + (j * 16) * 72 + kk * 32, 72);
        sacc[j] = __builtin_amdgcn_wmma_f32_16x16x32_f16(
            false, aQ[kk], false, bK, (short)0, sacc[j], false, false);
      }
    }

#pragma unroll
    for (int r = 0; r < 8; r++) {
      int qrow = qBase + r + 8 * hsel;
      float sv[4];
      float mv = -1e30f;
#pragma unroll
      for (int j = 0; j < 4; j++) {
        int key = kt * 64 + j * 16 + ln;
        float t = sacc[j][r] + slope * (float)(qrow - key);
        if (key > qrow) t = -1e30f;
        sv[j] = t;
        mv = fmaxf(mv, t);
      }
#pragma unroll
      for (int m = 8; m >= 1; m >>= 1) mv = fmaxf(mv, __shfl_xor(mv, m, 32));
      float mnew = fmaxf(m_arr[r], mv);
      float psum = 0.f;
#pragma unroll
      for (int j = 0; j < 4; j++) {
        float p = __expf(sv[j] - mnew);
        psum += p;
        Ps[(wave * 16 + r + 8 * hsel) * 72 + j * 16 + ln] = (_Float16)p;
      }
#pragma unroll
      for (int m = 8; m >= 1; m >>= 1) psum += __shfl_xor(psum, m, 32);
      float fac = __expf(m_arr[r] - mnew);
      l_arr[r] = l_arr[r] * fac + psum;
      m_arr[r] = mnew;
#pragma unroll
      for (int j = 0; j < 4; j++) oacc[j][r] *= fac;
    }
    // CDNA5 split wait: make per-wave P f16 stores visible before fragment reload
    asm volatile("s_wait_dscnt 0" ::: "memory");
#pragma unroll
    for (int kk = 0; kk < 2; kk++) {
      v16h aP = ldfrag(Ps + (wave * 16) * 72 + kk * 32, 72);
#pragma unroll
      for (int j = 0; j < 4; j++) {
        v16h bV = ldfrag(Vs + (j * 16) * 72 + kk * 32, 72);
        oacc[j] = __builtin_amdgcn_wmma_f32_16x16x32_f16(
            false, aP, false, bV, (short)0, oacc[j], false, false);
      }
    }
  }

#pragma unroll
  for (int j = 0; j < 4; j++) {
#pragma unroll
    for (int r = 0; r < 8; r++) {
      int qrow = qBase + r + 8 * hsel;
      int d = j * 16 + ln;
      float v = oacc[j][r] / l_arr[r];
      o16[((size_t)b * T_SEQ + qrow) * D_MODEL + h * HEADD + d] = (_Float16)v;
    }
  }
}

// ---------------------------------------------------------------------------
// Elementwise kernels
// ---------------------------------------------------------------------------
__global__ void __launch_bounds__(256)
rmsnorm_kernel(const float* __restrict__ x, const float* __restrict__ w,
               _Float16* __restrict__ out) {
  __shared__ float red[256];
  const int row = blockIdx.x, tid = threadIdx.x;
  const float* xr = x + (size_t)row * D_MODEL;
  float s = 0.f;
#pragma unroll
  for (int i = 0; i < 4; i++) { float v = xr[tid + i * 256]; s += v * v; }
  red[tid] = s;
  __syncthreads();
  for (int off = 128; off > 0; off >>= 1) {
    if (tid < off) red[tid] += red[tid + off];
    __syncthreads();
  }
  const float inv = rsqrtf(red[0] * (1.f / D_MODEL) + 1e-5f);
  _Float16* orow = out + (size_t)row * D_MODEL;
#pragma unroll
  for (int i = 0; i < 4; i++) {
    int c = tid + i * 256;
    orow[c] = (_Float16)(xr[c] * inv * w[c]);
  }
}

// Transpose + f32->f16: Wt[n*K+k] = W[k*N+n]
__global__ void __launch_bounds__(256)
convT(const float* __restrict__ W, _Float16* __restrict__ Wt, int K, int N) {
  size_t idx = (size_t)blockIdx.x * 256 + threadIdx.x;
  if (idx >= (size_t)K * N) return;
  int n = (int)(idx / K), k = (int)(idx % K);
  Wt[idx] = (_Float16)W[(size_t)k * N + n];
}

// Phase-rotate q (even/odd pairs), fold in 1/sqrt(HD), emit [B*H][T][64] f16.
__global__ void __launch_bounds__(256)
prep_q(const float* __restrict__ qlin, const float* __restrict__ qph,
       _Float16* __restrict__ q16) {
  size_t idx = (size_t)blockIdx.x * 256 + threadIdx.x;  // B*T*H*32 pairs
  int dp = idx & 31;
  int h  = (int)((idx >> 5) & 15);
  size_t bt = idx >> 9;
  int b = (int)(bt >> 10), t = (int)(bt & 1023);
  float ang = qph[h];
  float c = __cosf(ang), s = __sinf(ang);
  size_t src = bt * D_MODEL + h * HEADD + dp * 2;
  float q0 = qlin[src], q1 = qlin[src + 1];
  const float scale = 0.125f;  // 1/sqrt(64)
  size_t dst = ((size_t)(b * NHEAD + h) * T_SEQ + t) * HEADD + dp * 2;
  q16[dst]     = (_Float16)((q0 * c - q1 * s) * scale);
  q16[dst + 1] = (_Float16)((q0 * s + q1 * c) * scale);
}

// K = first KV_INNER channels of (unrotated) q, per reference.
__global__ void __launch_bounds__(256)
prep_k(const float* __restrict__ qlin, _Float16* __restrict__ k16) {
  size_t idx = (size_t)blockIdx.x * 256 + threadIdx.x;  // B*T*256
  int d = idx & 63;
  int g = (int)((idx >> 6) & 3);
  size_t bt = idx >> 8;
  int b = (int)(bt >> 10), t = (int)(bt & 1023);
  k16[((size_t)(b * KVHEADS + g) * T_SEQ + t) * HEADD + d] =
      (_Float16)qlin[bt * D_MODEL + g * HEADD + d];
}

// V transposed per kv-head: vt[b][g][d][t] so PV's B operand is Bt[N=hd][K=key].
__global__ void __launch_bounds__(256)
prep_v(const float* __restrict__ vlin, _Float16* __restrict__ vt16) {
  size_t idx = (size_t)blockIdx.x * 256 + threadIdx.x;  // B*T*256
  int d = idx & 63;
  int g = (int)((idx >> 6) & 3);
  size_t bt = idx >> 8;
  int b = (int)(bt >> 10), t = (int)(bt & 1023);
  vt16[((size_t)(b * KVHEADS + g) * HEADD + d) * T_SEQ + t] =
      (_Float16)vlin[bt * 256 + g * HEADD + d];
}

// ---------------------------------------------------------------------------
extern "C" void kernel_launch(void* const* d_in, const int* in_sizes, int n_in,
                              void* d_out, int out_size, void* d_ws, size_t ws_size,
                              hipStream_t stream) {
  const float* x     = (const float*)d_in[0];
  const float* ln1   = (const float*)d_in[1];
  const float* qW    = (const float*)d_in[2];
  const float* vW    = (const float*)d_in[3];
  const float* oW    = (const float*)d_in[4];
  const float* qph   = (const float*)d_in[5];
  const float* alibi = (const float*)d_in[6];
  const float* ln2   = (const float*)d_in[7];
  const float* fc1W  = (const float*)d_in[8];
  const float* fc1b  = (const float*)d_in[9];
  const float* fc2W  = (const float*)d_in[10];
  const float* fc2b  = (const float*)d_in[11];
  const float* lnf   = (const float*)d_in[12];
  const float* headW = (const float*)d_in[13];

  char* ws = (char*)d_ws;
  const size_t MB = 1ull << 20;
  float*    xw    = (float*)(ws + 0);          // 16 MB residual stream (f32)
  _Float16* h16   = (_Float16*)(ws + 16 * MB); //  8 MB rmsnorm output (f16)
  float*    qlin  = (float*)(ws + 24 * MB);    // 16 MB q projection (f32)
  float*    vlin  = (float*)(ws + 40 * MB);    //  4 MB v projection (f32)
  _Float16* q16   = (_Float16*)(ws + 44 * MB); //  8 MB rotated Q
  _Float16* k16   = (_Float16*)(ws + 52 * MB); //  2 MB K
  _Float16* vt16  = (_Float16*)(ws + 54 * MB); //  2 MB V^T
  _Float16* o16   = (_Float16*)(ws + 56 * MB); //  8 MB attn output
  _Float16* ffn16 = (_Float16*)(ws + 64 * MB); // 32 MB FFN hidden
  _Float16* wbuf  = (_Float16*)(ws + 96 * MB); //  8 MB transposed f16 weights

  (void)hipMemcpyAsync(xw, x, (size_t)M_ROWS * D_MODEL * sizeof(float),
                       hipMemcpyDeviceToDevice, stream);
  dim3 blk(256);

  for (int l = 0; l < 2; l++) {
    rmsnorm_kernel<<<M_ROWS, blk, 0, stream>>>(xw, ln1 + l * D_MODEL, h16);

    // q = x_pos @ qW[l]   (M=4096, K=512, N=1024); A = h16 cols [512:1024)
    convT<<<(512 * 1024) / 256, blk, 0, stream>>>(qW + (size_t)l * 512 * 1024, wbuf, 512, 1024);
    gemm_f16<false, false, false, false><<<dim3(8, 32), blk, 0, stream>>>(
        h16 + 512, D_MODEL, wbuf, 512, 1024, nullptr, nullptr, qlin, nullptr);
    // v = x_tok @ vW[l]   (K=512, N=256); A = h16 cols [0:512)
    convT<<<(512 * 256) / 256, blk, 0, stream>>>(vW + (size_t)l * 512 * 256, wbuf, 512, 256);
    gemm_f16<false, false, false, false><<<dim3(2, 32), blk, 0, stream>>>(
        h16, D_MODEL, wbuf, 512, 256, nullptr, nullptr, vlin, nullptr);

    prep_q<<<8192, blk, 0, stream>>>(qlin, qph + l * NHEAD, q16);
    prep_k<<<4096, blk, 0, stream>>>(qlin, k16);
    prep_v<<<4096, blk, 0, stream>>>(vlin, vt16);

    flash_attn<<<dim3(T_SEQ / 128, BATCH * NHEAD), blk, 0, stream>>>(
        q16, k16, vt16, alibi + l * NHEAD, o16);

    // x = x + o @ oW[l]   (K=1024, N=1024), residual add into xw
    convT<<<(1024 * 1024) / 256, blk, 0, stream>>>(oW + (size_t)l * 1024 * 1024, wbuf, 1024, 1024);
    gemm_f16<false, false, true, false><<<dim3(8, 32), blk, 0, stream>>>(
        o16, D_MODEL, wbuf, 1024, 1024, nullptr, xw, xw, nullptr);

    rmsnorm_kernel<<<M_ROWS, blk, 0, stream>>>(xw, ln2 + l * D_MODEL, h16);

    // ffn = gelu(h2 @ fc1_W + fc1_b)  (K=1024, N=4096) -> f16
    convT<<<(1024 * 4096) / 256, blk, 0, stream>>>(fc1W + (size_t)l * 1024 * 4096, wbuf, 1024, 4096);
    gemm_f16<true, true, false, true><<<dim3(32, 32), blk, 0, stream>>>(
        h16, D_MODEL, wbuf, 1024, 4096, fc1b + l * FFN_DIM, nullptr, nullptr, ffn16);
    // x = x + ffn @ fc2_W + fc2_b   (K=4096, N=1024)
    convT<<<(4096 * 1024) / 256, blk, 0, stream>>>(fc2W + (size_t)l * 4096 * 1024, wbuf, 4096, 1024);
    gemm_f16<true, false, true, false><<<dim3(8, 32), blk, 0, stream>>>(
        ffn16, FFN_DIM, wbuf, 4096, 1024, fc2b + l * D_MODEL, xw, xw, nullptr);
  }

  // out = rmsnorm(x, lnf) @ head_W   (K=1024, N=512) -> f32 d_out
  rmsnorm_kernel<<<M_ROWS, blk, 0, stream>>>(xw, lnf, h16);
  convT<<<(1024 * 512) / 256, blk, 0, stream>>>(headW, wbuf, 1024, 512);
  gemm_f16<false, false, false, false><<<dim3(4, 32), blk, 0, stream>>>(
      h16, D_MODEL, wbuf, 1024, 512, nullptr, nullptr, (float*)d_out, nullptr);
}
